// SelfAttentionHead_35476429865751
// MI455X (gfx1250) — matrix operand order repaired and checked by reading, not compiled, not run on hardware
//
#include <hip/hip_runtime.h>
#include <hip/hip_bf16.h>

// ---------------------------------------------------------------------------
// Causal self-attention head, MI455X (gfx1250), wave32 WMMA f16->f32.
// B=8, T=2048, C=1024, H=128.  Flash-attention with double-buffered
// async-to-LDS staging of K/V blocks (ASYNCcnt path) + WMMA everywhere.
// ---------------------------------------------------------------------------

#define B_ 8
#define T_ 2048
#define C_ 1024
#define H_ 128

typedef _Float16 v16h __attribute__((ext_vector_type(16)));
typedef _Float16 v8h  __attribute__((ext_vector_type(8)));
typedef float    v8f  __attribute__((ext_vector_type(8)));
typedef float    v4f  __attribute__((ext_vector_type(4)));
typedef int      v4i  __attribute__((ext_vector_type(4)));

// address-space-qualified pointer types matching the async builtin signature
typedef __attribute__((address_space(1))) v4i* gas_v4i_p;  // global
typedef __attribute__((address_space(3))) v4i* las_v4i_p;  // LDS

union V16 { v16h v; v8h h[2]; };

#ifndef __has_builtin
#define __has_builtin(x) 0
#endif

#if __has_builtin(__builtin_amdgcn_global_load_async_to_lds_b128)
#define HAVE_ASYNC_LDS 1
#else
#define HAVE_ASYNC_LDS 0
#endif

#if HAVE_ASYNC_LDS
// per-lane 16B DMA: global -> LDS, tracked by ASYNCcnt
#define ASYNC_COPY16(src, dst)                                                 \
  __builtin_amdgcn_global_load_async_to_lds_b128(                              \
      (gas_v4i_p)(src), (las_v4i_p)(dst), 0, 0)
#else
// fallback: global load + ds store (still pipelines across the barrier)
#define ASYNC_COPY16(src, dst)                                                 \
  do { *(v8h*)(dst) = *(const v8h*)(src); } while (0)
#endif

#if __has_builtin(__builtin_amdgcn_s_wait_asynccnt)
#define WAIT_ASYNC(N) __builtin_amdgcn_s_wait_asynccnt(N)
#elif HAVE_ASYNC_LDS
#define WAIT_ASYNC(N) asm volatile("s_wait_asynccnt %0" ::"i"(N) : "memory")
#else
#define WAIT_ASYNC(N) ((void)0)
#endif

__device__ __forceinline__ v8f wmma_f16(v16h a, v16h b, v8f c) {
  // D = A(16x32 f16) * B(32x16 f16) + C(16x16 f32)
  return __builtin_amdgcn_wmma_f32_16x16x32_f16(false, a, false, b,
                                                (short)0, c, false, false);
}

// ---------------------------------------------------------------------------
// Kernel 0: convert+transpose weights to f16.
// WT layout: [3][H][C], slot 0=Wq, 1=Wk, 2=Wv.  WpT[n][k] = Wp[k][n].
// ---------------------------------------------------------------------------
__global__ void cvt_weights_kernel(const float* __restrict__ W0,
                                   const float* __restrict__ W1,
                                   const float* __restrict__ W2,
                                   const float* __restrict__ Wp,
                                   _Float16* __restrict__ WT,
                                   _Float16* __restrict__ WpT) {
  int idx = blockIdx.x * blockDim.x + threadIdx.x;
  if (idx < 3 * C_ * H_) {
    int p = idx / (C_ * H_), r = idx % (C_ * H_);
    int c = r / H_, hh = r % H_;
    const float* W = (p == 0) ? W0 : (p == 1) ? W1 : W2;
    WT[(size_t)p * C_ * H_ + (size_t)hh * C_ + c] = (_Float16)W[(size_t)c * H_ + hh];
  } else {
    int r = idx - 3 * C_ * H_;
    int nn = r / H_, hh = r % H_;
    WpT[nn * H_ + hh] = (_Float16)Wp[hh * H_ + nn];
  }
}

// ---------------------------------------------------------------------------
// Kernel 1: fused QKV projection.  One wave = one 16x16 tile of one of q/k/v.
// q,k stored row-major f16 [B*T][H]; v stored transposed f16 [B][H][T].
// ---------------------------------------------------------------------------
__global__ void __launch_bounds__(256)
qkv_proj_kernel(const float* __restrict__ x, const _Float16* __restrict__ WT,
                _Float16* __restrict__ q16, _Float16* __restrict__ k16,
                _Float16* __restrict__ vT) {
  const int wid  = blockIdx.x * (blockDim.x >> 5) + (threadIdx.x >> 5);
  const int lane = threadIdx.x & 31;
  const int p    = wid >> 13;        // 8192 waves per projection
  const int rem  = wid & 8191;
  const int t0g  = (rem >> 3) << 4;  // global row (0..16383), tile of 16
  const int n0   = (rem & 7) << 4;   // H column tile
  const int h    = lane >> 4;
  const int m    = lane & 15;

  const _Float16* Wt   = WT + (size_t)p * (H_ * C_) + (size_t)(n0 + m) * C_;
  const float*    xrow = x + (size_t)(t0g + m) * C_;

  v8f acc = {};
  for (int c0 = 0; c0 < C_; c0 += 32) {
    const float* ax = xrow + c0 + h * 16;
    v4f f0 = *(const v4f*)(ax);
    v4f f1 = *(const v4f*)(ax + 4);
    v4f f2 = *(const v4f*)(ax + 8);
    v4f f3 = *(const v4f*)(ax + 12);
    V16 a;
#pragma unroll
    for (int i = 0; i < 4; ++i) {
      a.h[0][i]     = (_Float16)f0[i];
      a.h[0][i + 4] = (_Float16)f1[i];
      a.h[1][i]     = (_Float16)f2[i];
      a.h[1][i + 4] = (_Float16)f3[i];
    }
    v16h bw = *(const v16h*)(Wt + c0 + h * 16);
    acc = wmma_f16(a.v, bw, acc);
  }

  if (p == 2) {  // v -> transposed store [B][H][T]
    const int bb = t0g / T_, tl = t0g % T_;
    _Float16* dst = vT + ((size_t)bb * H_ + n0 + m) * T_ + tl + 8 * h;
#pragma unroll
    for (int r = 0; r < 8; ++r) dst[r] = (_Float16)acc[r];
  } else {       // q or k, row-major
    _Float16* dst = (p == 0) ? q16 : k16;
#pragma unroll
    for (int r = 0; r < 8; ++r)
      dst[(size_t)(t0g + 8 * h + r) * H_ + n0 + m] = (_Float16)acc[r];
  }
}

// ---------------------------------------------------------------------------
// Kernel 2: flash attention, 4 waves / workgroup, 64 queries / workgroup.
// K(32x128) and V(128x32) key-blocks are double-buffered in LDS and filled
// with async-to-LDS DMA; waves consume them with WMMA while the next block
// streams in.  Online softmax; fully-masked blocks degenerate to no-ops.
// ---------------------------------------------------------------------------
__global__ void __launch_bounds__(128)
attn_kernel(const _Float16* __restrict__ q16, const _Float16* __restrict__ k16,
            const _Float16* __restrict__ vT, _Float16* __restrict__ att) {
  __shared__ __align__(32) _Float16 Kbuf[2][32 * H_];   // 2 x 8KB
  __shared__ __align__(32) _Float16 Vbuf[2][H_ * 32];   // 2 x 8KB
  __shared__ __align__(32) _Float16 Plds[4][16 * 32];   // 1KB per wave

  const int tid  = threadIdx.x;
  const int wave = tid >> 5;
  const int lane = tid & 31;
  const int wg   = blockIdx.x;              // 0..255
  const int bb   = wg >> 5;                 // batch (32 WGs per batch)
  const int q0w  = ((wg & 31) << 6) + (wave << 4);  // this wave's query tile
  const int nb   = ((wg & 31) << 1) + 2;    // 32-key blocks for this WG
  const int h    = lane >> 4;
  const int n    = lane & 15;
  _Float16* P = Plds[wave];

  const _Float16* kg = k16 + (size_t)bb * T_ * H_;
  const _Float16* vg = vT + (size_t)bb * H_ * T_;

  // preload the wave's q A-operands (rows never change)
  const _Float16* qrow = q16 + (size_t)(bb * T_ + q0w + n) * H_;
  V16 qa[4];
#pragma unroll
  for (int t = 0; t < 4; ++t) {
    qa[t].h[0] = *(const v8h*)(qrow + t * 32 + h * 16);
    qa[t].h[1] = *(const v8h*)(qrow + t * 32 + h * 16 + 8);
  }

  // stage key-block j into buffer `buf`: 8 async b128 ops per wave
  auto stage = [&](int buf, int j) {
#pragma unroll
    for (int i = 0; i < 4; ++i) {           // K tile: 32 rows x 256B
      int c = tid + (i << 7);               // 0..511 chunks of 16B
      int row = c >> 4, off = (c & 15) << 3;
      ASYNC_COPY16(kg + (size_t)(j + row) * H_ + off, &Kbuf[buf][row * H_ + off]);
    }
#pragma unroll
    for (int i = 0; i < 4; ++i) {           // V tile: 128 rows x 64B
      int c = tid + (i << 7);
      int row = c >> 2, off = (c & 3) << 3;
      ASYNC_COPY16(vg + (size_t)row * T_ + j + off, &Vbuf[buf][row * 32 + off]);
    }
  };

  v8f zero = {};
  float Mx[8], L[8];
  v8f O[8];
#pragma unroll
  for (int r = 0; r < 8; ++r) { Mx[r] = -1e30f; L[r] = 0.f; O[r] = zero; }

  const float scale = 0.08838834764831845f;  // 1/sqrt(128)

  stage(0, 0);
  int cur = 0;
  for (int jb = 0; jb < nb; ++jb) {          // uniform across the workgroup
    const int j = jb << 5;
    if (jb + 1 < nb) {
      stage(cur ^ 1, j + 32);
      WAIT_ASYNC(8);                         // leave next block's 8 in flight
    } else {
      WAIT_ASYNC(0);
    }
    __syncthreads();                         // buf[cur] visible to all waves

    const _Float16* Kc = Kbuf[cur];
    const _Float16* Vc = Vbuf[cur];

    // ---- scores S[16 x 32] from LDS K tile
    v8f S0 = zero, S1 = zero;
#pragma unroll
    for (int t = 0; t < 4; ++t) {
      int hc = t * 32;
      v16h b0 = *(const v16h*)(Kc + n * H_ + hc + h * 16);
      v16h b1 = *(const v16h*)(Kc + (16 + n) * H_ + hc + h * 16);
      S0 = wmma_f16(qa[t].v, b0, S0);
      S1 = wmma_f16(qa[t].v, b1, S1);
    }
#pragma unroll
    for (int r = 0; r < 8; ++r) { S0[r] *= scale; S1[r] *= scale; }

    // ---- causal mask (wave-uniform branch; per-lane selects inside)
    if (j + 32 > q0w) {
#pragma unroll
      for (int r = 0; r < 8; ++r) {
        int qi = q0w + 8 * h + r;
        S0[r] = (j + n      <= qi) ? S0[r] : -1e30f;
        S1[r] = (j + 16 + n <= qi) ? S1[r] : -1e30f;
      }
    }

    // ---- online softmax: row stats via 16-lane xor reductions
    float p0[8], p1[8], al[8];
#pragma unroll
    for (int r = 0; r < 8; ++r) {
      float mx = fmaxf(S0[r], S1[r]);
#pragma unroll
      for (int d = 1; d < 16; d <<= 1) mx = fmaxf(mx, __shfl_xor(mx, d, 32));
      float newM = fmaxf(Mx[r], mx);
      al[r] = __expf(Mx[r] - newM);
      Mx[r] = newM;
      p0[r] = __expf(S0[r] - newM);
      p1[r] = __expf(S1[r] - newM);
      float sum = p0[r] + p1[r];
#pragma unroll
      for (int d = 1; d < 16; d <<= 1) sum += __shfl_xor(sum, d, 32);
      L[r] = L[r] * al[r] + sum;
    }
#pragma unroll
    for (int t = 0; t < 8; ++t)
#pragma unroll
      for (int r = 0; r < 8; ++r) O[t][r] *= al[r];

    // ---- P: D-layout f32 -> row-major f16 in LDS -> A-layout (wave-local)
#pragma unroll
    for (int r = 0; r < 8; ++r) {
      P[(8 * h + r) * 32 + n]      = (_Float16)p0[r];
      P[(8 * h + r) * 32 + 16 + n] = (_Float16)p1[r];
    }
    asm volatile("s_wait_dscnt 0" ::: "memory");  // cross-lane LDS RAW fence
    V16 pa;
    pa.h[0] = *(const v8h*)(P + n * 32 + h * 16);
    pa.h[1] = *(const v8h*)(P + n * 32 + h * 16 + 8);
    asm volatile("s_wait_dscnt 0" ::: "memory");  // WAR fence vs next iter

    // ---- O[16 x 128] += P[16 x 32] * V[32 x 128] from LDS V tile
#pragma unroll
    for (int t = 0; t < 8; ++t) {
      v16h bv = *(const v16h*)(Vc + (t * 16 + n) * 32 + h * 16);
      O[t] = wmma_f16(pa.v, bv, O[t]);
    }

    __syncthreads();                         // all waves done with buf[cur]
    cur ^= 1;
  }

  // ---- normalize and store attention output (f16, row-major [B*T][H])
  float inv[8];
#pragma unroll
  for (int r = 0; r < 8; ++r) inv[r] = 1.0f / L[r];
#pragma unroll
  for (int t = 0; t < 8; ++t)
#pragma unroll
    for (int r = 0; r < 8; ++r)
      att[(size_t)(bb * T_ + q0w + 8 * h + r) * H_ + t * 16 + n] =
          (_Float16)(O[t][r] * inv[r]);
}

// ---------------------------------------------------------------------------
// Kernel 3: out = att @ Wp + bp, f32 output.
// ---------------------------------------------------------------------------
__global__ void __launch_bounds__(256)
out_proj_kernel(const _Float16* __restrict__ att, const _Float16* __restrict__ WpT,
                const float* __restrict__ bp, float* __restrict__ out) {
  const int wid  = blockIdx.x * 8 + (threadIdx.x >> 5);
  const int lane = threadIdx.x & 31;
  const int row0 = (wid >> 3) << 4;
  const int n0   = (wid & 7) << 4;
  const int h = lane >> 4, m = lane & 15;

  const _Float16* arow = att + (size_t)(row0 + m) * H_;
  const _Float16* wrow = WpT + (size_t)(n0 + m) * H_;
  v8f acc = {};
#pragma unroll
  for (int hc = 0; hc < H_; hc += 32) {
    V16 a;
    a.h[0] = *(const v8h*)(arow + hc + h * 16);
    a.h[1] = *(const v8h*)(arow + hc + h * 16 + 8);
    v16h bw = *(const v16h*)(wrow + hc + h * 16);
    acc = wmma_f16(a.v, bw, acc);
  }
  float bias = bp[n0 + m];
#pragma unroll
  for (int r = 0; r < 8; ++r)
    out[(size_t)(row0 + 8 * h + r) * H_ + n0 + m] = acc[r] + bias;
}

// ---------------------------------------------------------------------------
extern "C" void kernel_launch(void* const* d_in, const int* in_sizes, int n_in,
                              void* d_out, int out_size, void* d_ws, size_t ws_size,
                              hipStream_t stream) {
  (void)in_sizes; (void)n_in; (void)out_size; (void)ws_size;
  const float* x  = (const float*)d_in[0];
  const float* Wk = (const float*)d_in[1];
  const float* Wq = (const float*)d_in[2];
  const float* Wv = (const float*)d_in[3];
  const float* Wp = (const float*)d_in[4];
  const float* bp = (const float*)d_in[5];
  float* out = (float*)d_out;

  const size_t BTH = (size_t)B_ * T_ * H_;   // 2,097,152
  _Float16* ws  = (_Float16*)d_ws;
  _Float16* q16 = ws;
  _Float16* k16 = q16 + BTH;
  _Float16* vT  = k16 + BTH;
  _Float16* att = vT + BTH;
  _Float16* WT  = att + BTH;                 // [3][H][C]
  _Float16* WpT = WT + 3 * (size_t)C_ * H_;  // [H][H]

  // 0: weights -> f16 transposed (409600 elements, exact grid)
  cvt_weights_kernel<<<(3 * C_ * H_ + H_ * H_) / 256, 256, 0, stream>>>(
      Wq, Wk, Wv, Wp, WT, WpT);
  // 1: QKV projection: 3 * 1024 row-tiles * 8 col-tiles = 24576 waves
  qkv_proj_kernel<<<24576 / 8, 256, 0, stream>>>(x, WT, q16, k16, vT);
  // 2: flash attention: 256 workgroups x 4 waves, 64 queries each
  attn_kernel<<<(B_ * T_) / 64, 128, 0, stream>>>(q16, k16, vT, att);
  // 3: output projection: 8192 waves
  out_proj_kernel<<<8192 / 8, 256, 0, stream>>>(att, WpT, bp, out);
}
